// AttentionOp_22497038697046
// MI455X (gfx1250) — matrix operand order repaired and checked by reading, not compiled, not run on hardware
//
#include <hip/hip_runtime.h>
#include <hip/hip_bf16.h>

typedef __attribute__((ext_vector_type(16))) _Float16 v16h;
typedef __attribute__((ext_vector_type(8)))  _Float16 v8h;
typedef __attribute__((ext_vector_type(2)))  _Float16 h2;
typedef __attribute__((ext_vector_type(8)))  float    v8f;
typedef __attribute__((ext_vector_type(4)))  float    v4f;

#define B_    4
#define S_    2048
#define NQ_   8
#define NKV_  2
#define D_    128
#define G_    (NQ_ / NKV_)
#define WIN_  1024
#define CAP_  50.0f
#define MASKV (-0.7f * 3.402823466e38f)

#define WAVES 4
#define QROWS (16 * WAVES)   // 64 q rows per block
#define KSTEP 32             // keys per iteration (= WMMA contraction depth for PV)

__device__ __forceinline__ h2 cvt2(float a, float b) {
    return __builtin_bit_cast(h2, __builtin_amdgcn_cvt_pkrtz(a, b));
}

__device__ __forceinline__ v8h pack8(v4f a, v4f b) {
    h2 c0 = cvt2(a[0], a[1]);
    h2 c1 = cvt2(a[2], a[3]);
    h2 c2 = cvt2(b[0], b[1]);
    h2 c3 = cvt2(b[2], b[3]);
    v8h r;
    r[0] = c0[0]; r[1] = c0[1]; r[2] = c1[0]; r[3] = c1[1];
    r[4] = c2[0]; r[5] = c2[1]; r[6] = c3[0]; r[7] = c3[1];
    return r;
}

__global__ __launch_bounds__(32 * WAVES)
void fa_gqa_swa_kernel(const float* __restrict__ q,
                       const float* __restrict__ k,
                       const float* __restrict__ v,
                       const int*   __restrict__ seg,
                       float* __restrict__ out)
{
    // K tile [32][128] f16, V tile transposed [128][32] f16, per-wave P [16][32] f16
    __shared__ _Float16 kbuf[KSTEP * D_];
    __shared__ _Float16 vbuf[D_ * KSTEP];
    __shared__ _Float16 pbuf[WAVES][16 * KSTEP];

    const int tid   = threadIdx.x;
    const int wave  = tid >> 5;
    const int lane  = tid & 31;
    const int halfw = lane >> 4;        // which 16-lane half
    const int ln    = lane & 15;
    const int base2 = halfw * 8;        // unused; A and B fragments both use kbase
    const int kbase = halfw * 16;       // A/B-fragment k offset (contiguous order k = 16h + s)

    const int qtiles = S_ / QROWS;
    const int bid = blockIdx.x;
    const int b   = bid / (NQ_ * qtiles);
    const int rem = bid % (NQ_ * qtiles);
    const int h   = rem / qtiles;
    const int q0  = (rem % qtiles) * QROWS;
    const int kvh = h / G_;             // GQA: q head h -> kv head h/G

    const int qrow0 = q0 + wave * 16;   // this wave's 16 q rows

    // ---- preload Q as 4 A-fragments (d chunks of 32), scaled by 1/sqrt(D), f32->f16 ----
    const float scale = 0.08838834764831845f; // 1/sqrt(128)
    v16h qa[4];
    {
        const float* qrow = q + (((size_t)b * S_ + (qrow0 + ln)) * NQ_ + h) * D_;
#pragma unroll
        for (int c = 0; c < 4; ++c) {
            const float* p0 = qrow + c * 32 + kbase;        // internal k = kbase+0..7
            const float* p1 = qrow + c * 32 + kbase + 8;    // internal k = kbase+8..15
            v4f a0 = *(const v4f*)(p0) * scale;
            v4f a1 = *(const v4f*)(p0 + 4) * scale;
            v4f b0 = *(const v4f*)(p1) * scale;
            v4f b1 = *(const v4f*)(p1 + 4) * scale;
            v8h lo = pack8(a0, a1);
            v8h hi = pack8(b0, b1);
            v16h t;
#pragma unroll
            for (int e = 0; e < 8; ++e) { t[e] = lo[e]; t[8 + e] = hi[e]; }
            qa[c] = t;
        }
    }

    // segment ids for my 8 C-layout rows (M = r + halfw*8)
    int segq[8];
#pragma unroll
    for (int r = 0; r < 8; ++r)
        segq[r] = seg[(size_t)b * S_ + qrow0 + r + halfw * 8];

    // online-softmax state: O accum (8 d-chunks of 16), row max (capped units), row sum
    v8f acc[8];
#pragma unroll
    for (int c = 0; c < 8; ++c) acc[c] = (v8f)0.f;
    float mrow[8], lrow[8], alpha[8], p0v[8], p1v[8];
#pragma unroll
    for (int r = 0; r < 8; ++r) { mrow[r] = MASKV; lrow[r] = 0.f; }

    const float L2E = 1.4426950408889634f;       // log2(e)
    const float KC  = -2.0f * L2E / CAP_;        // exp(-2x/CAP) == exp2(x*KC)

    // sliding-window key range for the whole block's 64 rows
    int lo = q0 - (WIN_ - 1); if (lo < 0) lo = 0; lo &= ~(KSTEP - 1);
    const int hi = q0 + QROWS - 1;

    for (int j0 = lo; j0 <= hi; j0 += KSTEP) {
        __syncthreads();   // previous iteration's LDS reads complete
        // ---- stage K row-major: thread owns a 32-wide d strip of one kv row ----
        {
            const int r  = tid >> 2;         // kv row 0..31
            const int d0 = (tid & 3) * 32;   // d chunk of 32
            const float* krow = k + (((size_t)b * S_ + (j0 + r)) * NKV_ + kvh) * D_ + d0;
#pragma unroll
            for (int t8 = 0; t8 < 4; ++t8) {
                v4f ka = *(const v4f*)(krow + t8 * 8);
                v4f kb = *(const v4f*)(krow + t8 * 8 + 4);
                *(v8h*)&kbuf[r * D_ + d0 + t8 * 8] = pack8(ka, kb);
            }
        }
        // ---- stage V transposed: thread owns one d column; lane-coalesced b32 reads,
        //      contiguous b128 LDS writes ----
        {
            const float* vcol = v + (((size_t)b * S_ + j0) * NKV_ + kvh) * D_ + tid;
            const int vstr = NKV_ * D_;
#pragma unroll
            for (int g = 0; g < 4; ++g) {
                v4f a, bb;
#pragma unroll
                for (int e = 0; e < 4; ++e) { a[e]  = vcol[(g * 8 + e) * vstr]; }
#pragma unroll
                for (int e = 0; e < 4; ++e) { bb[e] = vcol[(g * 8 + 4 + e) * vstr]; }
                *(v8h*)&vbuf[tid * KSTEP + g * 8] = pack8(a, bb);
            }
        }
        __syncthreads();

        // ---- S = Q K^T : two 16x16 logit tiles (kv cols 0-15 and 16-31) ----
        v8f s0 = (v8f)0.f, s1 = (v8f)0.f;
#pragma unroll
        for (int c = 0; c < 4; ++c) {
            const _Float16* kp0 = &kbuf[(0 * 16 + ln) * D_ + c * 32 + kbase];
            const _Float16* kp1 = &kbuf[(1 * 16 + ln) * D_ + c * 32 + kbase];
            v16h kf0, kf1;
            { v8h x0 = *(const v8h*)kp0; v8h x1 = *(const v8h*)(kp0 + 8);
#pragma unroll
              for (int e = 0; e < 8; ++e) { kf0[e] = x0[e]; kf0[8 + e] = x1[e]; } }
            { v8h x0 = *(const v8h*)kp1; v8h x1 = *(const v8h*)(kp1 + 8);
#pragma unroll
              for (int e = 0; e < 8; ++e) { kf1[e] = x0[e]; kf1[8 + e] = x1[e]; } }
            s0 = __builtin_amdgcn_wmma_f32_16x16x32_f16(false, qa[c], false, kf0, (short)0, s0, false, false);
            s1 = __builtin_amdgcn_wmma_f32_16x16x32_f16(false, qa[c], false, kf1, (short)0, s1, false, false);
        }

        // ---- soft-cap (exp2 + raw rcp), mask, online softmax ----
        const int seg_k0 = seg[(size_t)b * S_ + j0 + ln];
        const int seg_k1 = seg[(size_t)b * S_ + j0 + 16 + ln];
#pragma unroll
        for (int r = 0; r < 8; ++r) {
            const int i = qrow0 + r + halfw * 8;
            // tanh(x/CAP)*CAP = CAP*(2*rcp(1+exp2(x*KC)) - 1)
            float t0 = __builtin_amdgcn_exp2f(s0[r] * KC);
            float x0 = __builtin_fmaf(2.0f * CAP_, __builtin_amdgcn_rcpf(1.0f + t0), -CAP_);
            int j = j0 + ln;
            const bool ok0 = (j <= i) && ((i - j) < WIN_) && (segq[r] == seg_k0);
            x0 = ok0 ? x0 : MASKV;
            float t1 = __builtin_amdgcn_exp2f(s1[r] * KC);
            float x1 = __builtin_fmaf(2.0f * CAP_, __builtin_amdgcn_rcpf(1.0f + t1), -CAP_);
            j = j0 + 16 + ln;
            const bool ok1 = (j <= i) && ((i - j) < WIN_) && (segq[r] == seg_k1);
            x1 = ok1 ? x1 : MASKV;

            float mx = fmaxf(x0, x1);
#pragma unroll
            for (int off = 1; off < 16; off <<= 1) mx = fmaxf(mx, __shfl_xor(mx, off, 32));
            const float mnew = fmaxf(mrow[r], mx);
            const float al = __builtin_amdgcn_exp2f((mrow[r] - mnew) * L2E);
            const float e0 = ok0 ? __builtin_amdgcn_exp2f((x0 - mnew) * L2E) : 0.f;
            const float e1 = ok1 ? __builtin_amdgcn_exp2f((x1 - mnew) * L2E) : 0.f;
            float sum = e0 + e1;
#pragma unroll
            for (int off = 1; off < 16; off <<= 1) sum += __shfl_xor(sum, off, 32);
            lrow[r] = lrow[r] * al + sum;
            mrow[r] = mnew;
            alpha[r] = al;
            p0v[r] = e0;
            p1v[r] = e1;
        }

        // rescale O accumulators by alpha
#pragma unroll
        for (int c = 0; c < 8; ++c)
#pragma unroll
            for (int r = 0; r < 8; ++r) acc[c][r] *= alpha[r];

        // ---- P: C layout -> A layout via per-wave LDS (same-wave DS ops are in-order) ----
        _Float16* pb = pbuf[wave];
#pragma unroll
        for (int r = 0; r < 8; ++r) {
            pb[(r + halfw * 8) * KSTEP + ln]      = (_Float16)p0v[r];
            pb[(r + halfw * 8) * KSTEP + 16 + ln] = (_Float16)p1v[r];
        }
        v16h pa;
        {
            const _Float16* pr = &pb[ln * KSTEP];
            v8h a0 = *(const v8h*)(pr + kbase);
            v8h a1 = *(const v8h*)(pr + kbase + 8);
#pragma unroll
            for (int e = 0; e < 8; ++e) { pa[e] = a0[e]; pa[8 + e] = a1[e]; }
        }

        // ---- O += P V : 8 d-chunks of 16 ----
#pragma unroll
        for (int c = 0; c < 8; ++c) {
            const _Float16* vp = &vbuf[(c * 16 + ln) * KSTEP + kbase];
            v16h vf;
            { v8h x0 = *(const v8h*)vp; v8h x1 = *(const v8h*)(vp + 8);
#pragma unroll
              for (int e = 0; e < 8; ++e) { vf[e] = x0[e]; vf[8 + e] = x1[e]; } }
            acc[c] = __builtin_amdgcn_wmma_f32_16x16x32_f16(false, pa, false, vf, (short)0, acc[c], false, false);
        }
    }

    // ---- epilogue: out = O / l ----
#pragma unroll
    for (int r = 0; r < 8; ++r) {
        const int i = qrow0 + r + halfw * 8;
        const float inv = lrow[r] > 0.f ? 1.f / lrow[r] : 0.f;
        float* orow = out + (((size_t)b * S_ + i) * NQ_ + h) * D_;
#pragma unroll
        for (int c = 0; c < 8; ++c) orow[c * 16 + ln] = acc[c][r] * inv;
    }
}

extern "C" void kernel_launch(void* const* d_in, const int* in_sizes, int n_in,
                              void* d_out, int out_size, void* d_ws, size_t ws_size,
                              hipStream_t stream) {
    (void)in_sizes; (void)n_in; (void)out_size; (void)d_ws; (void)ws_size;
    const float* q  = (const float*)d_in[0];
    const float* k  = (const float*)d_in[1];
    const float* v  = (const float*)d_in[2];
    const int*  seg = (const int*)d_in[3];
    float* out = (float*)d_out;
    dim3 grid(B_ * NQ_ * (S_ / QROWS));
    dim3 block(32 * WAVES);
    hipLaunchKernelGGL(fa_gqa_swa_kernel, grid, block, 0, stream, q, k, v, seg, out);
}
